// BahdanauAttention_72705206386813
// MI455X (gfx1250) — matrix operand order repaired and checked by reading, not compiled, not run on hardware
//
#include <hip/hip_runtime.h>
#include <hip/hip_bf16.h>
#include <math.h>

// ---------------- problem constants ----------------
#define HDIM 1024
#define SDIM 2048
#define BDIM 32

// ---------------- WMMA types (CDNA5 / gfx1250, wave32) ----------------
typedef __attribute__((ext_vector_type(16))) __bf16 v16bf;
typedef __attribute__((ext_vector_type(8)))  float  v8f;

union Frag { v16bf v; uint4 u[2]; };

// round-to-nearest-even f32 -> bf16 bits
__device__ __forceinline__ unsigned f2bf(float f) {
    unsigned u = __float_as_uint(f);
    return (u + 0x7FFFu + ((u >> 16) & 1u)) >> 16;
}

// =====================================================================
// Kernel 1: Ua_w (f32, [H][H]) -> bf16 workspace copy
// =====================================================================
__global__ __launch_bounds__(256) void ua_to_bf16_kernel(
    const float* __restrict__ Ua, unsigned short* __restrict__ out)
{
    int idx = blockIdx.x * 256 + threadIdx.x;      // one uint4 (8 bf16) per thread
    const float4* src = (const float4*)Ua;
    float4 a = src[idx * 2 + 0];
    float4 c = src[idx * 2 + 1];
    uint4 p;
    p.x = f2bf(a.x) | (f2bf(a.y) << 16);
    p.y = f2bf(a.z) | (f2bf(a.w) << 16);
    p.z = f2bf(c.x) | (f2bf(c.y) << 16);
    p.w = f2bf(c.z) | (f2bf(c.w) << 16);
    ((uint4*)out)[idx] = p;
}

// =====================================================================
// Kernel 2: qq[b][o] = dot(ht[b], Wa_w[o]) + Wa_b[o] + Ua_b[o]
// =====================================================================
__global__ __launch_bounds__(256) void qq_kernel(
    const float* __restrict__ ht, const float* __restrict__ Wa,
    const float* __restrict__ Wa_b, const float* __restrict__ Ua_b,
    float* __restrict__ qq)
{
    __shared__ float sht[HDIM];
    const int b = blockIdx.x, tid = threadIdx.x;
    for (int i = tid; i < HDIM / 4; i += 256)
        ((float4*)sht)[i] = ((const float4*)(ht + (size_t)b * HDIM))[i];
    __syncthreads();
    const int o = blockIdx.y * 256 + tid;
    const float* wrow = Wa + (size_t)o * HDIM;
    float acc = 0.f;
    for (int h = 0; h < HDIM; h += 4) {
        float4 wv = *(const float4*)(wrow + h);
        acc += sht[h] * wv.x + sht[h + 1] * wv.y + sht[h + 2] * wv.z + sht[h + 3] * wv.w;
    }
    qq[(size_t)b * HDIM + o] = acc + Wa_b[o] + Ua_b[o];
}

// =====================================================================
// Kernel 3 (main): fused  k = hs@Ua^T -> tanh(k+qq) -> dot with Va
//
// Block = (batch b) x (128 s-rows).  8 waves, wave w owns M-tile w.
// LDS holds A (hs tile, bf16) and B (Ua tile, bf16) in *fragment order*
//   region layout: [tile(32KB)] [kt(1KB)] [inner(512B)] [lane(16B)]
// so every ds_load_b128 spans all 64 banks (minimal 2-cycle) and the
// total footprint is exactly 256KB + 64KB = 320KB (CDNA5 per-WG max).
//
// Each wave register-blocks 2 N-tiles (acc0/acc1 share one A fragment:
// 6 ds_load_b128 per 2 wmma) and software-pipelines fragments (double
// buffer) so LDS latency overlaps the WMMA pipe.
// B staging uses CDNA5 GLOBAL_LOAD_ASYNC_TO_LDS_B128 (per-lane LDS
// destination VGPR = free swizzle), tracked with ASYNCcnt.
// =====================================================================
#define BM        128
#define A_BYTES   (BM * HDIM * 2)            // 262144
#define B_OFF     A_BYTES
#define LDS_BYTES (A_BYTES + 32 * HDIM * 2)  // 327680 = 320 KB exactly

__global__ __launch_bounds__(256) void fused_score_kernel(
    const float* __restrict__ hs,
    const unsigned short* __restrict__ UaB,   // Ua in bf16, [H][H]
    const float* __restrict__ qq,             // [B][H]
    const float* __restrict__ Va,             // [H]
    const float* __restrict__ Va_b,           // [1]
    float* __restrict__ scores)               // [B][S]
{
    extern __shared__ __align__(16) char smem[];
    const int tid = threadIdx.x;
    const int b   = blockIdx.y;
    const int s0  = blockIdx.x * BM;

    // ---- stage A: hs[b, s0..s0+127, :] f32 -> bf16, fragment order ----
    // unit = 16B of LDS (8 bf16) = 32B of global (2 float4)
    // batched 4 units/iter so 8 global_load_b128 are in flight per wave
    {
        const float* src = hs + ((size_t)b * SDIM + s0) * HDIM;
        for (int it = 0; it < 64; it += 4) {          // 64 units per thread
            float4 f[8];
            const int idx0 = tid + it * 256;
#pragma unroll
            for (int u = 0; u < 4; ++u) {
                int idx = idx0 + u * 256;
                int r = idx >> 7, c = idx & 127;      // 128 units per row
                const float4* p = (const float4*)(src + (size_t)r * HDIM + c * 8);
                f[u * 2 + 0] = p[0];
                f[u * 2 + 1] = p[1];
            }
#pragma unroll
            for (int u = 0; u < 4; ++u) {
                int idx = idx0 + u * 256;
                int r = idx >> 7, c = idx & 127;
                int mt = r >> 4, lr2 = r & 15;
                int kt = c >> 2, hf = c & 1, inner = (c >> 1) & 1;
                int ln = hf * 16 + lr2;
                uint4 pk;
                pk.x = f2bf(f[u*2].x)   | (f2bf(f[u*2].y)   << 16);
                pk.y = f2bf(f[u*2].z)   | (f2bf(f[u*2].w)   << 16);
                pk.z = f2bf(f[u*2+1].x) | (f2bf(f[u*2+1].y) << 16);
                pk.w = f2bf(f[u*2+1].z) | (f2bf(f[u*2+1].w) << 16);
                *(uint4*)(smem + mt * 32768 + kt * 1024 + inner * 512 + ln * 16) = pk;
            }
        }
    }
    __syncthreads();

    const int w    = tid >> 5;        // wave id 0..7 (wave32): M-tile
    const int lane = tid & 31;
    const int half = lane >> 4;
    const int lr   = lane & 15;

    const char* Abase  = smem + w * 32768 + lane * 16;
    const char* B0base = smem + B_OFF + lane * 16;
    const char* B1base = smem + B_OFF + 32768 + lane * 16;

    float scoreAcc[8];
#pragma unroll
    for (int v = 0; v < 8; ++v) scoreAcc[v] = 0.f;

    for (int ro = 0; ro < HDIM / 32; ++ro) {          // 32 rounds, 2 N-tiles each
        __syncthreads();                              // prior B fully consumed
        // ---- stage B: Ua_bf16 rows [ro*32, ro*32+32) via async DMA ----
        {
            const unsigned short* bs = UaB + (size_t)(ro * 32) * HDIM;
            for (int i = 0; i < 16; ++i) {            // 4096 16B units / 256 thr
                int idx = tid + i * 256;
                int r = idx >> 7, c = idx & 127;
                int t = r >> 4, o_in = r & 15;
                int kt = c >> 2, hf = c & 1, inner = (c >> 1) & 1;
                unsigned lds_off = B_OFF + t * 32768 + kt * 1024 + inner * 512
                                 + (hf * 16 + o_in) * 16;
                unsigned goff = (unsigned)(r * (HDIM * 2) + c * 16);
                asm volatile("global_load_async_to_lds_b128 %0, %1, %2"
                             :: "v"(lds_off), "v"(goff), "s"(bs) : "memory");
            }
            asm volatile("s_wait_asynccnt 0x0" ::: "memory");
        }
        __syncthreads();

        v8f acc0 = {}, acc1 = {};
        Frag fa[2], fb0[2], fb1[2];
        // prologue: fragments for kt=0
        fa[0].u[0]  = *(const uint4*)(Abase);
        fa[0].u[1]  = *(const uint4*)(Abase + 512);
        fb0[0].u[0] = *(const uint4*)(B0base);
        fb0[0].u[1] = *(const uint4*)(B0base + 512);
        fb1[0].u[0] = *(const uint4*)(B1base);
        fb1[0].u[1] = *(const uint4*)(B1base + 512);
#pragma unroll
        for (int kt = 0; kt < HDIM / 32; ++kt) {      // 32 K-steps
            const int cur = kt & 1, nxt = cur ^ 1;
            if (kt < HDIM / 32 - 1) {                 // prefetch kt+1
                const int off = (kt + 1) * 1024;
                fa[nxt].u[0]  = *(const uint4*)(Abase + off);
                fa[nxt].u[1]  = *(const uint4*)(Abase + off + 512);
                fb0[nxt].u[0] = *(const uint4*)(B0base + off);
                fb0[nxt].u[1] = *(const uint4*)(B0base + off + 512);
                fb1[nxt].u[0] = *(const uint4*)(B1base + off);
                fb1[nxt].u[1] = *(const uint4*)(B1base + off + 512);
            }
            acc0 = __builtin_amdgcn_wmma_f32_16x16x32_bf16(
                false, fa[cur].v, false, fb0[cur].v, (short)0, acc0, false, false);
            acc1 = __builtin_amdgcn_wmma_f32_16x16x32_bf16(
                false, fa[cur].v, false, fb1[cur].v, (short)0, acc1, false, false);
        }

        // epilogue: e = tanh(k + qq); score += e * Va   (C/D: N=lane%16, M=v+8*half)
        const int oa = ro * 32 + lr;
        const int ob = ro * 32 + 16 + lr;
        const float qa = qq[(size_t)b * HDIM + oa], va_a = Va[oa];
        const float qb = qq[(size_t)b * HDIM + ob], va_b = Va[ob];
#pragma unroll
        for (int v = 0; v < 8; ++v) {
            scoreAcc[v] += tanhf(acc0[v] + qa) * va_a;
            scoreAcc[v] += tanhf(acc1[v] + qb) * va_b;
        }
    }

    // ---- reduce per-row score across lanes/waves (reuse B region) ----
    __syncthreads();
    float* sscore = (float*)(smem + B_OFF);
    if (tid < BM) sscore[tid] = 0.f;
    __syncthreads();
#pragma unroll
    for (int v = 0; v < 8; ++v)
        atomicAdd(&sscore[w * 16 + v + 8 * half], scoreAcc[v]);
    __syncthreads();
    if (tid < BM)
        scores[(size_t)b * SDIM + s0 + tid] = sscore[tid] + Va_b[0];
}

// =====================================================================
// Kernel 4: softmax over S per batch, write attn
// =====================================================================
__global__ __launch_bounds__(256) void softmax_kernel(
    const float* __restrict__ scores, float* __restrict__ attn)
{
    __shared__ float red[256];
    const int b = blockIdx.x, tid = threadIdx.x;
    const float* row = scores + (size_t)b * SDIM;
    float v[8];
    float m = -INFINITY;
#pragma unroll
    for (int i = 0; i < 8; ++i) { v[i] = row[tid + i * 256]; m = fmaxf(m, v[i]); }
    red[tid] = m; __syncthreads();
    for (int s = 128; s > 0; s >>= 1) {
        if (tid < s) red[tid] = fmaxf(red[tid], red[tid + s]);
        __syncthreads();
    }
    m = red[0]; __syncthreads();
    float sum = 0.f;
#pragma unroll
    for (int i = 0; i < 8; ++i) { v[i] = __expf(v[i] - m); sum += v[i]; }
    red[tid] = sum; __syncthreads();
    for (int s = 128; s > 0; s >>= 1) {
        if (tid < s) red[tid] += red[tid + s];
        __syncthreads();
    }
    const float inv = 1.f / red[0];
#pragma unroll
    for (int i = 0; i < 8; ++i)
        attn[(size_t)b * SDIM + tid + i * 256] = v[i] * inv;
}

// =====================================================================
// Kernel 5: ctx[b][h] = sum_s attn[b][s] * hs[b][s][h]  (2nd hs pass)
// =====================================================================
__global__ __launch_bounds__(256) void ctx_kernel(
    const float* __restrict__ hs, const float* __restrict__ attn,
    float* __restrict__ ctx)
{
    __shared__ float sa[SDIM];
    const int b = blockIdx.y, tid = threadIdx.x;
    for (int i = tid; i < SDIM / 4; i += 256)
        ((float4*)sa)[i] = ((const float4*)(attn + (size_t)b * SDIM))[i];
    __syncthreads();
    const int h = blockIdx.x * 256 + tid;
    const float* base = hs + (size_t)b * SDIM * HDIM + h;
    float acc = 0.f;
#pragma unroll 8
    for (int s = 0; s < SDIM; ++s)
        acc += sa[s] * base[(size_t)s * HDIM];
    ctx[(size_t)b * HDIM + h] = acc;
}

// =====================================================================
// launch
// =====================================================================
extern "C" void kernel_launch(void* const* d_in, const int* in_sizes, int n_in,
                              void* d_out, int out_size, void* d_ws, size_t ws_size,
                              hipStream_t stream) {
    const float* ht   = (const float*)d_in[0];   // [B,1,H]
    const float* hs   = (const float*)d_in[1];   // [B,S,H]
    const float* Wa_w = (const float*)d_in[2];   // [H,H]
    const float* Wa_b = (const float*)d_in[3];   // [H]
    const float* Ua_w = (const float*)d_in[4];   // [H,H]
    const float* Ua_b = (const float*)d_in[5];   // [H]
    const float* Va_w = (const float*)d_in[6];   // [1,H]
    const float* Va_b = (const float*)d_in[7];   // [1]

    float* out_ctx  = (float*)d_out;                         // [B,H]
    float* out_attn = (float*)d_out + (size_t)BDIM * HDIM;   // [B,S]

    // workspace layout
    char* ws = (char*)d_ws;
    unsigned short* UaB = (unsigned short*)ws;                        // 2 MB
    float* qq     = (float*)(ws + (size_t)HDIM * HDIM * 2);           // 128 KB
    float* scores = (float*)(ws + (size_t)HDIM * HDIM * 2 + (size_t)BDIM * HDIM * 4);

    // 1) Ua -> bf16
    ua_to_bf16_kernel<<<(HDIM * HDIM / 8) / 256, 256, 0, stream>>>(Ua_w, UaB);

    // 2) qq = ht@Wa^T + Wa_b + Ua_b
    qq_kernel<<<dim3(BDIM, HDIM / 256), 256, 0, stream>>>(ht, Wa_w, Wa_b, Ua_b, qq);

    // 3) fused WMMA GEMM + tanh + Va reduction -> scores
    (void)hipFuncSetAttribute((const void*)fused_score_kernel,
                              hipFuncAttributeMaxDynamicSharedMemorySize, LDS_BYTES);
    fused_score_kernel<<<dim3(SDIM / BM, BDIM), 256, LDS_BYTES, stream>>>(
        hs, UaB, qq, Va_w, Va_b, scores);

    // 4) softmax -> attn (second half of d_out)
    softmax_kernel<<<BDIM, 256, 0, stream>>>(scores, out_attn);

    // 5) ctx = attn @ hs (first part of d_out)
    ctx_kernel<<<dim3(HDIM / 256, BDIM), 256, 0, stream>>>(hs, out_attn, out_ctx);
}
